// MemoryEfficientAttention_49194555408463
// MI455X (gfx1250) — matrix operand order repaired and checked by reading, not compiled, not run on hardware
//
#include <hip/hip_runtime.h>

// ---------------------------------------------------------------------------
// MI455X (gfx1250) WMMA implementation of GroupNorm + MHSA + LayerNorm block.
// All matmuls use v_wmma_f32_16x16x32_f16. Fragments load as 2 x ds_load_b128.
// GEMM and attention are software-pipelined with double-buffered LDS so global
// loads of tile i+1 overlap WMMAs of tile i (one barrier per step).
// ---------------------------------------------------------------------------

typedef __attribute__((ext_vector_type(16))) _Float16 v16h;
typedef __attribute__((ext_vector_type(8)))  float    v8f;

#define Cc   512
#define Gg   32
#define NHh  8
#define HDd  64
#define Bb   4
#define Hh   48
#define Ww   48
#define Ss   (Hh*Ww)       // 2304
#define BSs  (Bb*Ss)       // 9216
#define EPSf 1e-5f

union AFrag { v16h v; _Float16 h[16]; uint4 q[2]; };
union Acc   { v8f  v; float    f[8];  };
union H8    { uint4 u; _Float16 h[8]; };

// A-matrix 16x32 f16 k order used in this file: lane = {m = lane%16, hw = lane/16}.
// element e -> K = e + 16*hw: one contiguous 16-half run at
// row-offset (16*hw) -> 2 x b128 per fragment.
// B-matrix 32x16: element e -> K = e + 16*hw, N = lane%16: contiguous 16-half
// run when the tile is stored N-major ([n][k]) -> 2 x b128 per fragment.

// ---------------------------------------------------------------------------
// 1) fp32 -> f16 conversion (weights)
// ---------------------------------------------------------------------------
__global__ void f32_to_f16_kernel(const float* __restrict__ src,
                                  _Float16* __restrict__ dst, int n) {
  int i = blockIdx.x * blockDim.x + threadIdx.x;
  if (i < n) dst[i] = (_Float16)src[i];
}

// ---------------------------------------------------------------------------
// 2a) GroupNorm statistics: one block per (batch, group)
// ---------------------------------------------------------------------------
__global__ void __launch_bounds__(256)
gn_stats_kernel(const float* __restrict__ x, float* __restrict__ stats) {
  const int b = blockIdx.x / Gg, g = blockIdx.x % Gg;
  const int CPG = Cc / Gg;                 // 16 channels per group
  const int cnt = CPG * Ss;                // 36864 elements
  float s = 0.f, ss = 0.f;
  for (int i = threadIdx.x; i < cnt; i += 256) {
    int c  = g * CPG + i / Ss;
    int sp = i % Ss;
    float v = x[((size_t)b * Cc + c) * Ss + sp];
    s += v; ss += v * v;
  }
  __shared__ float rs[256], rss[256];
  rs[threadIdx.x] = s; rss[threadIdx.x] = ss;
  __syncthreads();
  for (int o = 128; o > 0; o >>= 1) {
    if (threadIdx.x < o) { rs[threadIdx.x] += rs[threadIdx.x + o];
                           rss[threadIdx.x] += rss[threadIdx.x + o]; }
    __syncthreads();
  }
  if (threadIdx.x == 0) {
    float mu  = rs[0] / cnt;
    float var = rss[0] / cnt - mu * mu;
    stats[blockIdx.x * 2 + 0] = mu;
    stats[blockIdx.x * 2 + 1] = rsqrtf(var + EPSf);
  }
}

// ---------------------------------------------------------------------------
// 2b) GroupNorm apply + NCHW -> [B, S, C] transpose, f16 output
// ---------------------------------------------------------------------------
__global__ void gn_apply_kernel(const float* __restrict__ x,
                                const float* __restrict__ stats,
                                const float* __restrict__ gamma,
                                const float* __restrict__ beta,
                                _Float16* __restrict__ h16) {
  size_t i = (size_t)blockIdx.x * blockDim.x + threadIdx.x;   // over B*C*S
  if (i >= (size_t)Bb * Cc * Ss) return;
  int sp = (int)(i % Ss);
  int c  = (int)((i / Ss) % Cc);
  int b  = (int)(i / ((size_t)Cc * Ss));
  int g  = c / (Cc / Gg);
  float mu   = stats[(b * Gg + g) * 2 + 0];
  float rstd = stats[(b * Gg + g) * 2 + 1];
  float v = (x[i] - mu) * rstd * gamma[c] + beta[c];
  h16[((size_t)b * Ss + sp) * Cc + c] = (_Float16)v;
}

// ---------------------------------------------------------------------------
// 3) Tiled WMMA GEMM: out[M,N] = A[M,K] @ W[K,N] + bias
//    128x64 block tile, 8 waves x (16x64 stripe). Double-buffered LDS with
//    register-staged prefetch: global loads of K-step i+1 overlap WMMAs of i.
// ---------------------------------------------------------------------------
__global__ void __launch_bounds__(256)
wmma_gemm_kernel(const _Float16* __restrict__ A, const _Float16* __restrict__ Wm,
                 const float* __restrict__ bias,
                 _Float16* __restrict__ out16, float* __restrict__ out32,
                 int M, int N, int K) {
  __shared__ __align__(16) union {
    struct { _Float16 As[2][128 * 32]; _Float16 Bt[2][64 * 32]; } s; // 24KB
    _Float16 Ot[8][16 * 64];                                          // 16KB
  } sh;
  const int tid  = threadIdx.x;
  const int wave = tid >> 5, lane = tid & 31;
  const int m0 = blockIdx.x * 128, n0 = blockIdx.y * 64;
  const int mr = lane & 15, hw = lane >> 4;
  const int wk = tid >> 3, wng = tid & 7;          // W-tile staging coords
  const int ar = tid >> 2, aq4 = tid & 3;          // A-tile staging coords

  Acc acc[4];
  #pragma unroll
  for (int t = 0; t < 4; ++t)
    #pragma unroll
    for (int j = 0; j < 8; ++j) acc[t].f[j] = 0.f;

  // --- pipelined staging helpers (all b128 global, b128/b16 LDS) -----------
  const _Float16* aSrc0 = A + (size_t)(m0 + ar) * K + aq4 * 8;        // it=0
  const _Float16* aSrc1 = A + (size_t)(m0 + 64 + ar) * K + aq4 * 8;   // it=1
  const _Float16* wSrc  = Wm + (size_t)wk * N + n0 + wng * 8;

  uint4 aR0 = *(const uint4*)(aSrc0);
  uint4 aR1 = *(const uint4*)(aSrc1);
  H8 wR; wR.u = *(const uint4*)(wSrc);
  {
    ((uint4*)sh.s.As[0])[ar * 4 + aq4]        = aR0;
    ((uint4*)sh.s.As[0])[(64 + ar) * 4 + aq4] = aR1;
    #pragma unroll
    for (int j = 0; j < 8; ++j) sh.s.Bt[0][(wng * 8 + j) * 32 + wk] = wR.h[j];
  }

  const int nk = K >> 5;
  for (int i = 0; i < nk; ++i) {
    __syncthreads();
    const int cur = i & 1, nxt = cur ^ 1;
    if (i + 1 < nk) {                       // issue next tile's global loads
      aR0 = *(const uint4*)(aSrc0 + (i + 1) * 32);
      aR1 = *(const uint4*)(aSrc1 + (i + 1) * 32);
      wR.u = *(const uint4*)(wSrc + (size_t)(i + 1) * 32 * N);
    }
    if (i + 2 < nk) {                       // L2 prefetch two tiles ahead
      __builtin_prefetch(aSrc0 + (i + 2) * 32, 0, 1);
      __builtin_prefetch(wSrc + (size_t)(i + 2) * 32 * N, 0, 1);
    }

    const _Float16* AsC = sh.s.As[cur];
    const _Float16* BtC = sh.s.Bt[cur];
    AFrag af;
    af.q[0] = *(const uint4*)(AsC + (wave * 16 + mr) * 32 + 16 * hw);
    af.q[1] = *(const uint4*)(AsC + (wave * 16 + mr) * 32 + 16 * hw + 8);
    AFrag bf[4];
    #pragma unroll
    for (int nt = 0; nt < 4; ++nt) {
      bf[nt].q[0] = *(const uint4*)(BtC + (nt * 16 + mr) * 32 + 16 * hw);
      bf[nt].q[1] = *(const uint4*)(BtC + (nt * 16 + mr) * 32 + 16 * hw + 8);
    }
    #pragma unroll
    for (int nt = 0; nt < 4; ++nt)
      acc[nt].v = __builtin_amdgcn_wmma_f32_16x16x32_f16(
          false, af.v, false, bf[nt].v, (short)0, acc[nt].v, false, false);

    if (i + 1 < nk) {                       // fill the other buffer
      ((uint4*)sh.s.As[nxt])[ar * 4 + aq4]        = aR0;
      ((uint4*)sh.s.As[nxt])[(64 + ar) * 4 + aq4] = aR1;
      #pragma unroll
      for (int j = 0; j < 8; ++j)
        sh.s.Bt[nxt][(wng * 8 + j) * 32 + wk] = wR.h[j];
    }
  }
  __syncthreads();   // LDS reused as output staging below

  // Epilogue. C layout: lane (v) -> (M = v + 8*hw, N = nt*16 + lane%16).
  if (out16) {
    _Float16* OtW = sh.Ot[wave];
    #pragma unroll
    for (int nt = 0; nt < 4; ++nt) {
      float bv = bias[n0 + nt * 16 + mr];
      #pragma unroll
      for (int v = 0; v < 8; ++v)
        OtW[(v + 8 * hw) * 64 + nt * 16 + mr] = (_Float16)(acc[nt].f[v] + bv);
    }
    // Per-wave region + in-order DS ops: no barrier needed.
    #pragma unroll
    for (int i = 0; i < 4; ++i) {
      int u = lane * 4 + i;              // 0..127 uint4 chunks of 16x64 tile
      int row = u >> 3, c8 = u & 7;
      *(uint4*)(out16 + (size_t)(m0 + wave * 16 + row) * N + n0 + c8 * 8) =
          ((const uint4*)OtW)[u];
    }
  } else {
    #pragma unroll
    for (int nt = 0; nt < 4; ++nt) {
      int col = n0 + nt * 16 + mr;
      float bv = bias[col];
      #pragma unroll
      for (int v = 0; v < 8; ++v)
        out32[(size_t)(m0 + wave * 16 + v + 8 * hw) * N + col] =
            acc[nt].f[v] + bv;
    }
  }
}

// ---------------------------------------------------------------------------
// 4) Flash attention. Block = 8 waves = 128 queries of one (batch, head).
//    Double-buffered K/V chunks (32 keys): K row-major [key][d], V transposed
//    [d][key] -> all B-fragments are 2 x ds_load_b128. Next chunk's global
//    loads overlap this chunk's 8 WMMAs + online softmax.
// ---------------------------------------------------------------------------
__global__ void __launch_bounds__(256)
wmma_attn_kernel(const _Float16* __restrict__ q16, const _Float16* __restrict__ k16,
                 const _Float16* __restrict__ v16p, _Float16* __restrict__ o16) {
  __shared__ __align__(16) _Float16 Kt[2][32 * 64];    // [key][d]    8KB
  __shared__ __align__(16) _Float16 Vt[2][64 * 32];    // [d][key]    8KB
  __shared__ __align__(16) _Float16 Pso[8][16 * 64];   // per-wave   16KB

  const int nqb = Ss / 128;                            // 18 query blocks
  const int blk  = blockIdx.x;
  const int b    = blk / (NHh * nqb);
  const int head = (blk / nqb) % NHh;
  const int qb   = blk % nqb;
  const int tid  = threadIdx.x;
  const int wave = tid >> 5, lane = tid & 31;
  const int mr = lane & 15, hw = lane >> 4;
  const int qrow0 = qb * 128 + wave * 16;              // this wave's 16 queries
  const size_t base = ((size_t)b * Ss) * Cc + head * HDd;

  // Q fragments: 16 queries x 64 dims = two K=32 A-fragments, 4 b128 loads.
  AFrag aq[2];
  #pragma unroll
  for (int dc = 0; dc < 2; ++dc) {
    const _Float16* qrow = q16 + base + (size_t)(qrow0 + mr) * Cc + dc * 32;
    aq[dc].q[0] = *(const uint4*)(qrow + 16 * hw);
    aq[dc].q[1] = *(const uint4*)(qrow + 16 * hw + 8);
  }

  Acc o[4];
  float mrow[8], lrow[8];
  #pragma unroll
  for (int t = 0; t < 4; ++t)
    #pragma unroll
    for (int j = 0; j < 8; ++j) o[t].f[j] = 0.f;
  #pragma unroll
  for (int v = 0; v < 8; ++v) { mrow[v] = -1e30f; lrow[v] = 0.f; }

  _Float16* PsW = Pso[wave];

  // --- pipelined K/V staging ------------------------------------------------
  const int skey = tid >> 3, sdg = tid & 7;
  const _Float16* kSrc = k16 + base + (size_t)skey * Cc + sdg * 8;
  const _Float16* vSrc = v16p + base + (size_t)skey * Cc + sdg * 8;

  uint4 kR = *(const uint4*)(kSrc);
  H8 vR; vR.u = *(const uint4*)(vSrc);
  {
    ((uint4*)Kt[0])[tid] = kR;
    #pragma unroll
    for (int j = 0; j < 8; ++j) Vt[0][(sdg * 8 + j) * 32 + skey] = vR.h[j];
  }

  const int nch = Ss / 32;                 // 72 key chunks
  for (int i = 0; i < nch; ++i) {
    __syncthreads();
    const int cur = i & 1, nxt = cur ^ 1;
    if (i + 1 < nch) {                     // next chunk's global loads
      kR = *(const uint4*)(kSrc + (size_t)(i + 1) * 32 * Cc);
      vR.u = *(const uint4*)(vSrc + (size_t)(i + 1) * 32 * Cc);
    }
    if (i + 2 < nch) {
      __builtin_prefetch(kSrc + (size_t)(i + 2) * 32 * Cc, 0, 1);
      __builtin_prefetch(vSrc + (size_t)(i + 2) * 32 * Cc, 0, 1);
    }
    const _Float16* KtC = Kt[cur];
    const _Float16* VtC = Vt[cur];

    // ---- scores = (Q @ K^T) * scale over the 32-key chunk ------------------
    AFrag bk[2][2];
    #pragma unroll
    for (int dc = 0; dc < 2; ++dc) {
      const _Float16* kr0 = KtC + (size_t)mr * 64 + dc * 32 + 16 * hw;
      const _Float16* kr1 = KtC + (size_t)(16 + mr) * 64 + dc * 32 + 16 * hw;
      bk[dc][0].q[0] = *(const uint4*)(kr0);
      bk[dc][0].q[1] = *(const uint4*)(kr0 + 8);
      bk[dc][1].q[0] = *(const uint4*)(kr1);
      bk[dc][1].q[1] = *(const uint4*)(kr1 + 8);
    }
    Acc c0, c1;
    #pragma unroll
    for (int j = 0; j < 8; ++j) { c0.f[j] = 0.f; c1.f[j] = 0.f; }
    c0.v = __builtin_amdgcn_wmma_f32_16x16x32_f16(
        false, aq[0].v, false, bk[0][0].v, (short)0, c0.v, false, false);
    c1.v = __builtin_amdgcn_wmma_f32_16x16x32_f16(
        false, aq[0].v, false, bk[0][1].v, (short)0, c1.v, false, false);
    c0.v = __builtin_amdgcn_wmma_f32_16x16x32_f16(
        false, aq[1].v, false, bk[1][0].v, (short)0, c0.v, false, false);
    c1.v = __builtin_amdgcn_wmma_f32_16x16x32_f16(
        false, aq[1].v, false, bk[1][1].v, (short)0, c1.v, false, false);

    // ---- online softmax (row lives in one 16-lane group) -------------------
    float alpha[8];
    #pragma unroll
    for (int v = 0; v < 8; ++v) {
      float s0 = c0.f[v] * 0.125f;      // 1/sqrt(64)
      float s1 = c1.f[v] * 0.125f;
      float tm = fmaxf(s0, s1);
      #pragma unroll
      for (int msk = 1; msk < 16; msk <<= 1)
        tm = fmaxf(tm, __shfl_xor(tm, msk, 16));
      float nm = fmaxf(mrow[v], tm);
      float a  = __expf(mrow[v] - nm);
      float p0 = __expf(s0 - nm), p1 = __expf(s1 - nm);
      float rs = p0 + p1;
      #pragma unroll
      for (int msk = 1; msk < 16; msk <<= 1)
        rs += __shfl_xor(rs, msk, 16);
      lrow[v] = lrow[v] * a + rs;
      mrow[v] = nm;
      alpha[v] = a;
      c0.f[v] = p0; c1.f[v] = p1;
    }
    #pragma unroll
    for (int t = 0; t < 4; ++t)
      #pragma unroll
      for (int v = 0; v < 8; ++v) o[t].f[v] *= alpha[v];

    // ---- P (C layout) -> per-wave LDS 16x32 row-major -> A-fragment --------
    #pragma unroll
    for (int v = 0; v < 8; ++v) {
      int Mr = v + 8 * hw;
      PsW[Mr * 32 + mr]      = (_Float16)c0.f[v];
      PsW[Mr * 32 + 16 + mr] = (_Float16)c1.f[v];
    }
    // Wave-local region; DS ops are in-order within a wave -> no barrier.
    AFrag ap;
    ap.q[0] = *(const uint4*)(PsW + mr * 32 + 16 * hw);
    ap.q[1] = *(const uint4*)(PsW + mr * 32 + 16 * hw + 8);

    // ---- O += P @ V (V^T staged: contiguous fragment rows) -----------------
    AFrag bv[4];
    #pragma unroll
    for (int nt = 0; nt < 4; ++nt) {
      const _Float16* vr = VtC + (size_t)(nt * 16 + mr) * 32 + 16 * hw;
      bv[nt].q[0] = *(const uint4*)(vr);
      bv[nt].q[1] = *(const uint4*)(vr + 8);
    }
    #pragma unroll
    for (int nt = 0; nt < 4; ++nt)
      o[nt].v = __builtin_amdgcn_wmma_f32_16x16x32_f16(
          false, ap.v, false, bv[nt].v, (short)0, o[nt].v, false, false);

    if (i + 1 < nch) {                     // fill the other K/V buffer
      ((uint4*)Kt[nxt])[tid] = kR;
      #pragma unroll
      for (int j = 0; j < 8; ++j) Vt[nxt][(sdg * 8 + j) * 32 + skey] = vR.h[j];
    }
  }

  // ---- finalize: /rowsum, transpose via per-wave LDS, b128 stores ----------
  #pragma unroll
  for (int nt = 0; nt < 4; ++nt)
    #pragma unroll
    for (int v = 0; v < 8; ++v)
      PsW[(v + 8 * hw) * 64 + nt * 16 + mr] = (_Float16)(o[nt].f[v] / lrow[v]);
  #pragma unroll
  for (int i = 0; i < 4; ++i) {
    int u = lane * 4 + i;                 // 0..127 uint4 chunks of 16x64 tile
    int row = u >> 3, c8 = u & 7;
    *(uint4*)(o16 + base + (size_t)(qrow0 + row) * Cc + c8 * 8) =
        ((const uint4*)PsW)[u];
  }
}

// ---------------------------------------------------------------------------
// 5) LayerNorm over C + transpose [B,S,C] -> [B,C,H,W] + residual add
// ---------------------------------------------------------------------------
__global__ void __launch_bounds__(128)
ln_residual_kernel(const float* __restrict__ y, const float* __restrict__ xres,
                   const float* __restrict__ g, const float* __restrict__ bta,
                   float* __restrict__ out) {
  int r = blockIdx.x;                      // 0..BS-1
  int b = r / Ss, sp = r % Ss;
  const float* yr = y + (size_t)r * Cc;
  float s = 0.f, ss = 0.f;
  for (int c = threadIdx.x; c < Cc; c += 128) {
    float v = yr[c]; s += v; ss += v * v;
  }
  __shared__ float rs[128], rss[128];
  rs[threadIdx.x] = s; rss[threadIdx.x] = ss;
  __syncthreads();
  for (int o = 64; o > 0; o >>= 1) {
    if (threadIdx.x < o) { rs[threadIdx.x] += rs[threadIdx.x + o];
                           rss[threadIdx.x] += rss[threadIdx.x + o]; }
    __syncthreads();
  }
  float mu   = rs[0] / Cc;
  float rstd = rsqrtf(rss[0] / Cc - mu * mu + EPSf);
  for (int c = threadIdx.x; c < Cc; c += 128) {
    float v = (yr[c] - mu) * rstd * g[c] + bta[c];
    size_t oi = ((size_t)b * Cc + c) * Ss + sp;
    out[oi] = v + xres[oi];                // RESCALE == 1.0
  }
}

// ---------------------------------------------------------------------------
// Host-side launch
// ---------------------------------------------------------------------------
extern "C" void kernel_launch(void* const* d_in, const int* in_sizes, int n_in,
                              void* d_out, int out_size, void* d_ws, size_t ws_size,
                              hipStream_t stream) {
  (void)in_sizes; (void)n_in; (void)out_size; (void)ws_size;
  const float* x   = (const float*)d_in[0];
  const float* gng = (const float*)d_in[1];
  const float* gnb = (const float*)d_in[2];
  const float* wq  = (const float*)d_in[3];
  const float* bq  = (const float*)d_in[4];
  const float* wk  = (const float*)d_in[5];
  const float* bk  = (const float*)d_in[6];
  const float* wv  = (const float*)d_in[7];
  const float* bv  = (const float*)d_in[8];
  const float* wo  = (const float*)d_in[9];
  const float* bo  = (const float*)d_in[10];
  const float* lng = (const float*)d_in[11];
  const float* lnb = (const float*)d_in[12];
  float* out = (float*)d_out;

  // Workspace carve-out (~68 MB total), 256B aligned slices.
  char* ws = (char*)d_ws;
  size_t off = 0;
  auto alloc = [&](size_t bytes) {
    char* p = ws + off;
    off += (bytes + 255) & ~(size_t)255;
    return p;
  };
  float*    stats  = (float*)   alloc((size_t)Bb * Gg * 2 * sizeof(float));
  _Float16* wq16   = (_Float16*)alloc((size_t)Cc * Cc * 2);
  _Float16* wk16   = (_Float16*)alloc((size_t)Cc * Cc * 2);
  _Float16* wv16   = (_Float16*)alloc((size_t)Cc * Cc * 2);
  _Float16* wo16   = (_Float16*)alloc((size_t)Cc * Cc * 2);
  _Float16* h16    = (_Float16*)alloc((size_t)BSs * Cc * 2);
  _Float16* q16    = (_Float16*)alloc((size_t)BSs * Cc * 2);
  _Float16* k16    = (_Float16*)alloc((size_t)BSs * Cc * 2);
  _Float16* v16    = (_Float16*)alloc((size_t)BSs * Cc * 2);
  _Float16* attn16 = (_Float16*)alloc((size_t)BSs * Cc * 2);
  float*    y32    = (float*)   alloc((size_t)BSs * Cc * sizeof(float));

  const int nW = Cc * Cc;
  f32_to_f16_kernel<<<(nW + 255) / 256, 256, 0, stream>>>(wq, wq16, nW);
  f32_to_f16_kernel<<<(nW + 255) / 256, 256, 0, stream>>>(wk, wk16, nW);
  f32_to_f16_kernel<<<(nW + 255) / 256, 256, 0, stream>>>(wv, wv16, nW);
  f32_to_f16_kernel<<<(nW + 255) / 256, 256, 0, stream>>>(wo, wo16, nW);

  gn_stats_kernel<<<Bb * Gg, 256, 0, stream>>>(x, stats);
  {
    size_t n = (size_t)Bb * Cc * Ss;
    gn_apply_kernel<<<(unsigned)((n + 255) / 256), 256, 0, stream>>>(
        x, stats, gng, gnb, h16);
  }

  dim3 ggrid(BSs / 128, Cc / 64);
  wmma_gemm_kernel<<<ggrid, 256, 0, stream>>>(h16, wq16, bq, q16, nullptr,
                                              BSs, Cc, Cc);
  wmma_gemm_kernel<<<ggrid, 256, 0, stream>>>(h16, wk16, bk, k16, nullptr,
                                              BSs, Cc, Cc);
  wmma_gemm_kernel<<<ggrid, 256, 0, stream>>>(h16, wv16, bv, v16, nullptr,
                                              BSs, Cc, Cc);

  wmma_attn_kernel<<<Bb * NHh * (Ss / 128), 256, 0, stream>>>(q16, k16, v16,
                                                              attn16);

  wmma_gemm_kernel<<<ggrid, 256, 0, stream>>>(attn16, wo16, bo, nullptr, y32,
                                              BSs, Cc, Cc);

  ln_residual_kernel<<<BSs, 128, 0, stream>>>(y32, x, lng, lnb, out);
}
